// _LibXLSTM_Mixed_6640019439857
// MI455X (gfx1250) — compile-verified, matrix operand order checked by reading, not run on hardware
//
#include <hip/hip_runtime.h>

// ---------------- problem constants ----------------
#define BQ     512
#define SQ     64
#define DQ     128
#define VQ     256
#define NHQ    4
#define HDQ    32
#define INNERQ 256
#define NHMQ   4
#define DHMQ   64
#define KCONV  4
#define NBQ    64
#define BSQ    4
#define FFQ    128
#define MROWS  (BQ*SQ)     // 32768
#define BHQ    (BQ*NHMQ)   // 2048

typedef __attribute__((ext_vector_type(16))) _Float16 v16h;
typedef __attribute__((ext_vector_type(8)))  float    v8f;
typedef __attribute__((ext_vector_type(4)))  int      v4i;

#if defined(__HIP_DEVICE_COMPILE__) && __has_builtin(__builtin_amdgcn_global_load_async_to_lds_b128)
#define USE_ASYNC_LDS 1
#else
#define USE_ASYNC_LDS 0
#endif

__device__ __forceinline__ void wait_async0(){
#if defined(__HIP_DEVICE_COMPILE__)
#if __has_builtin(__builtin_amdgcn_s_wait_asynccnt)
  __builtin_amdgcn_s_wait_asynccnt(0);
#else
  asm volatile("s_wait_asynccnt 0" ::: "memory");
#endif
#endif
}

// ---------------- device helpers ----------------
__device__ __forceinline__ float sigm_(float x){ return 1.f/(1.f+expf(-x)); }
__device__ __forceinline__ float silu_(float x){ return x*sigm_(x); }
__device__ __forceinline__ float logsig_(float x){ return fminf(x,0.f) - log1pf(expf(-fabsf(x))); }
__device__ __forceinline__ float gelu_(float x){
  float x3 = x*x*x;
  return 0.5f*x*(1.f+tanhf(0.7978845608028654f*(x+0.044715f*x3)));
}

// pack 8 contiguous f32 (two aligned float4s) into halves dst[o..o+7]
__device__ __forceinline__ void cvt8(v16h& dst, int o, const float* __restrict__ p){
  float4 x0 = *(const float4*)(p);
  float4 x1 = *(const float4*)(p+4);
  dst[o+0]=(_Float16)x0.x; dst[o+1]=(_Float16)x0.y; dst[o+2]=(_Float16)x0.z; dst[o+3]=(_Float16)x0.w;
  dst[o+4]=(_Float16)x1.x; dst[o+5]=(_Float16)x1.y; dst[o+6]=(_Float16)x1.z; dst[o+7]=(_Float16)x1.w;
}

// ---------------- embedding ----------------
__global__ void embed_kernel(const int* __restrict__ x, const float* __restrict__ embed,
                             float* __restrict__ H){
  int row = blockIdx.x, t = threadIdx.x;
  int idx = x[row];
  H[(size_t)row*DQ + t] = embed[(size_t)idx*DQ + t];
}

// ---------------- rowwise LayerNorm (D=128) ----------------
__global__ void ln_kernel(const float* __restrict__ X, const float* __restrict__ w,
                          float* __restrict__ Y){
  int row = blockIdx.x, t = threadIdx.x;
  __shared__ float sh[DQ];
  float v = X[(size_t)row*DQ + t];
  sh[t]=v; __syncthreads();
  for (int st=64; st>0; st>>=1){ if (t<st) sh[t]+=sh[t+st]; __syncthreads(); }
  float mu = sh[0]*(1.f/DQ); __syncthreads();
  float d = v-mu; sh[t]=d*d; __syncthreads();
  for (int st=64; st>0; st>>=1){ if (t<st) sh[t]+=sh[t+st]; __syncthreads(); }
  float var = sh[0]*(1.f/DQ);
  Y[(size_t)row*DQ + t] = d*rsqrtf(var+1e-5f)*w[t];
}

// ---------------- WMMA GEMM: C = A[MxK] @ W[KxN] (+bias)(+res) ----------------
// Block = 4 waves sharing one 16-wide N tile; W tile staged in LDS (async copy),
// each wave owns one 16-row M tile. Kd <= 256, Kd%32==0, N%16==0, M%64==0.
template<bool HASB, bool HASR>
__global__ void gemm_wmma_kernel(const float* __restrict__ A, const float* __restrict__ W,
                                 const float* __restrict__ bias, const float* __restrict__ res,
                                 float* __restrict__ C, int M, int N, int Kd){
  __shared__ __align__(16) float wtile[256*16];   // 16KB max (K up to 256)
  int t = threadIdx.x;
  int lane = t & 31, wave = t >> 5;
  int nt = blockIdx.x;
  int colL = lane & 15;
  // ---- stage W[:, nt*16 .. nt*16+15] into LDS ----
  {
    const float* gbase = W + (size_t)nt*16;
    int units = Kd*4;                       // 16B units
#if USE_ASYNC_LDS
    for (int u=t; u<units; u+=128){
      int k = u>>2, q = (u&3)*4;
      __builtin_amdgcn_global_load_async_to_lds_b128(
          (__attribute__((address_space(1))) v4i*)(gbase + (size_t)k*N + q),
          (__attribute__((address_space(3))) v4i*)(&wtile[k*16 + q]), 0, 0);
    }
    wait_async0();
#else
    for (int u=t; u<units; u+=128){
      int k = u>>2, q = (u&3)*4;
      *(float4*)&wtile[k*16 + q] = *(const float4*)(gbase + (size_t)k*N + q);
    }
#endif
    __syncthreads();
  }
  int mt = blockIdx.y*4 + wave;
  int row_a = mt*16 + colL;
  v8f acc = {};
  for (int k0=0; k0<Kd; k0+=32){
    int ko = k0 + ((lane&16)?8:0);
    const float* abase = A + (size_t)row_a*Kd + ko;
    __builtin_prefetch((const void*)(abase + 32), 0, 1);
    v16h a, b;
    cvt8(a, 0, abase);
    cvt8(a, 8, abase + 16);
    const float* lb = &wtile[ko*16 + colL];
#pragma unroll
    for (int e=0;e<8;e++) b[e]   = (_Float16)lb[e*16];
#pragma unroll
    for (int e=0;e<8;e++) b[8+e] = (_Float16)lb[(16+e)*16];
    acc = __builtin_amdgcn_wmma_f32_16x16x32_f16(false, a, false, b, (short)0, acc, false, false);
  }
  int rbase = mt*16 + ((lane&16)?8:0);
  int col   = nt*16 + colL;
  float bv = 0.f;
  if constexpr (HASB) bv = bias[col];
#pragma unroll
  for (int r=0;r<8;r++){
    float v = acc[r] + bv;
    if constexpr (HASR) v += res[(size_t)(rbase+r)*N + col];
    C[(size_t)(rbase+r)*N + col] = v;
  }
}

// ---------------- sLSTM gate pre-projection (block-diagonal, per-head) ----------------
__global__ void slstm_gx_kernel(const float* __restrict__ XS, const float* __restrict__ gw,
                                const float* __restrict__ bias, float* __restrict__ GX){
  long id = (long)blockIdx.x*256 + threadIdx.x;
  int e = id & 31; int n = (id>>5)&3; int g = (id>>7)&3; long bs = id>>9;
  float acc = bias[(g*NHQ+n)*HDQ + e];
  const float* xr = XS + bs*DQ + n*HDQ;
  const float* wr = gw + ((size_t)((g*NHQ+n)*HDQ + e))*HDQ;
#pragma unroll
  for (int d=0; d<HDQ; d++) acc += xr[d]*wr[d];
  GX[id] = acc;
}

// ---------------- sLSTM recurrent scan: one block per (b, head) ----------------
__global__ void slstm_scan_kernel(const float* __restrict__ GX, const float* __restrict__ rw,
                                  float* __restrict__ Y){
  int bn = blockIdx.x; int b = bn >> 2; int n = bn & 3;
  int t = threadIdx.x; int g = t >> 5; int e = t & 31;
  __shared__ float rws[4*HDQ*HDQ];
  __shared__ float hs[HDQ], cs_[HDQ], ns_[HDQ], ms_[HDQ];
  __shared__ float raws[4][HDQ];
  for (int i=t; i<4*HDQ*HDQ; i+=128){
    int d = i & 31; int e2 = (i>>5)&31; int g2 = i>>10;
    rws[i] = rw[((size_t)((g2*NHQ+n)*HDQ + e2))*HDQ + d];
  }
  if (t < HDQ){ hs[t]=0.f; cs_[t]=0.f; ns_[t]=0.f; ms_[t]=0.f; }
  __syncthreads();
  const float* rwrow = &rws[(g*HDQ + e)*HDQ];
  for (int s=0; s<SQ; s++){
    float acc = GX[((long)(b*SQ+s))*512 + g*DQ + n*HDQ + e];
#pragma unroll
    for (int d=0; d<HDQ; d++) acc += hs[d]*rwrow[d];
    raws[g][e] = acc;
    __syncthreads();
    if (g==0){
      float ir=raws[0][e], fr=raws[1][e], zr=raws[2][e], orr=raws[3][e];
      float lfm = ms_[e] + logsig_(fr);
      float mn  = fmaxf(ir, lfm);
      float i   = expf(ir - mn);
      float f   = expf(lfm - mn);
      float cn  = f*cs_[e] + i*tanhf(zr);
      float nn  = f*ns_[e] + i;
      float hn  = sigm_(orr)*cn/nn;
      cs_[e]=cn; ns_[e]=nn; ms_[e]=mn; hs[e]=hn;
      Y[((long)(b*SQ+s))*DQ + n*HDQ + e] = hn;
    }
    __syncthreads();
  }
}

// ---------------- per-head LN, scale, add into residual H (width 128) ----------------
__global__ void mhln_add_kernel(const float* __restrict__ Yin, const float* __restrict__ gn,
                                float* __restrict__ H, int hd){
  int row = blockIdx.x, t = threadIdx.x;
  __shared__ float sh[DQ];
  float v = Yin[(size_t)row*DQ + t];
  int seg = t & ~(hd-1); int lane = t & (hd-1);
  sh[t]=v; __syncthreads();
  for (int st=hd>>1; st>0; st>>=1){ if (lane<st) sh[t]+=sh[t+st]; __syncthreads(); }
  float mu = sh[seg]/(float)hd; __syncthreads();
  float d = v-mu; sh[t]=d*d; __syncthreads();
  for (int st=hd>>1; st>0; st>>=1){ if (lane<st) sh[t]+=sh[t+st]; __syncthreads(); }
  float var = sh[seg]/(float)hd;
  H[(size_t)row*DQ + t] += d*rsqrtf(var+1e-5f)*gn[t];
}

// ---------------- FFN gate: t2 = gelu(g)*u ----------------
__global__ void gelu_gate_kernel(const float* __restrict__ F1, float* __restrict__ F2){
  long id = (long)blockIdx.x*256 + threadIdx.x;
  long row = id / FFQ; int f = id % FFQ;
  float g = F1[row*(2*FFQ) + f];
  float u = F1[row*(2*FFQ) + FFQ + f];
  F2[id] = gelu_(g)*u;
}

// ---------------- causal depthwise conv (K=4) + silu ----------------
__global__ void conv_kernel(const float* __restrict__ U, const float* __restrict__ cw,
                            const float* __restrict__ cb, float* __restrict__ XC){
  long id = (long)blockIdx.x*256 + threadIdx.x;
  int c = id & 255; long bs = id >> 8; int s = bs % SQ; long b = bs / SQ;
  float acc = cb[c];
#pragma unroll
  for (int k=0;k<KCONV;k++){
    int si = s - (KCONV-1) + k;
    if (si >= 0) acc += U[((b*SQ+si))*(2*INNERQ) + c] * cw[c*KCONV + k];
  }
  XC[id] = silu_(acc);
}

// ---------------- block-diagonal qkv ----------------
__global__ void qkv_kernel(const float* __restrict__ XC, const float* __restrict__ U,
                           const float* __restrict__ qkvw,
                           float* __restrict__ Qo, float* __restrict__ Ko, float* __restrict__ Vo){
  long id = (long)blockIdx.x*256 + threadIdx.x;
  int t = blockIdx.y;
  int c = id & 255; long row = id >> 8;
  int nb = c >> 2; int o = c & 3;
  const float* w = qkvw + (((size_t)t*NBQ + nb)*BSQ + o)*BSQ;
  float acc = 0.f;
#pragma unroll
  for (int i=0;i<BSQ;i++){
    float xin = (t==2) ? U[row*(2*INNERQ) + nb*4 + i] : XC[row*INNERQ + nb*4 + i];
    acc += xin * w[i];
  }
  if (t==0) Qo[id]=acc; else if (t==1) Ko[id]=acc; else Vo[id]=acc;
}

// ---------------- input/forget gate projections ----------------
__global__ void gates_kernel(const float* __restrict__ Q, const float* __restrict__ K,
                             const float* __restrict__ V,
                             const float* __restrict__ igw, const float* __restrict__ igb,
                             const float* __restrict__ fgw, const float* __restrict__ fgb,
                             float* __restrict__ IG, float* __restrict__ FG){
  long id = (long)blockIdx.x*128 + threadIdx.x;
  int h = id & 3; long row = id >> 2;
  float ai = igb[h], af = fgb[h];
  const float* qrow = Q + row*INNERQ;
  const float* krow = K + row*INNERQ;
  const float* vrow = V + row*INNERQ;
  for (int c=0;c<INNERQ;c++){ float g=qrow[c]; ai += g*igw[(c          )*NHMQ+h]; af += g*fgw[(c          )*NHMQ+h]; }
  for (int c=0;c<INNERQ;c++){ float g=krow[c]; ai += g*igw[(INNERQ+c   )*NHMQ+h]; af += g*fgw[(INNERQ+c   )*NHMQ+h]; }
  for (int c=0;c<INNERQ;c++){ float g=vrow[c]; ai += g*igw[(2*INNERQ+c )*NHMQ+h]; af += g*fgw[(2*INNERQ+c )*NHMQ+h]; }
  long b = row / SQ; int s = row % SQ;
  IG[(b*NHMQ+h)*SQ + s] = ai;
  FG[(b*NHMQ+h)*SQ + s] = af;
}

// ---------------- sequential cumsum of log_sigmoid(fg) ----------------
__global__ void cumsum_kernel(const float* __restrict__ FG, float* __restrict__ CS){
  int bh = blockIdx.x;
  if (threadIdx.x==0){
    float acc=0.f; CS[bh*(SQ+1)] = 0.f;
    for (int s=0;s<SQ;s++){ acc += logsig_(FG[bh*SQ+s]); CS[bh*(SQ+1)+s+1]=acc; }
  }
}

// ---------------- batched qk: qk[i][j] = (q_i . k_j)/8 ----------------
__global__ void attn_qk_kernel(const float* __restrict__ Q, const float* __restrict__ K,
                               float* __restrict__ QK){
  int lane = threadIdx.x & 31;
  int t = blockIdx.x;  int bh = blockIdx.y;
  int b = bh >> 2; int h = bh & 3;
  int mt = t >> 2, nt = t & 3;
  int rowbase = b*SQ; int hoff = h*DHMQ;
  int i_a = mt*16 + (lane&15);
  int j_b = nt*16 + (lane&15);
  v8f acc = {};
  for (int k0=0; k0<DHMQ; k0+=32){
    int ka = k0 + ((lane&16)?8:0);
    const float* qb = Q + (size_t)(rowbase+i_a)*INNERQ + hoff + ka;
    const float* kb = K + (size_t)(rowbase+j_b)*INNERQ + hoff + ka;   // kT: contiguous in d
    v16h a, bv;
    cvt8(a, 0, qb);  cvt8(a, 8, qb+16);
    cvt8(bv, 0, kb); cvt8(bv, 8, kb+16);
    acc = __builtin_amdgcn_wmma_f32_16x16x32_f16(false, a, false, bv, (short)0, acc, false, false);
  }
  int ib = mt*16 + ((lane&16)?8:0);
  int j  = nt*16 + (lane&15);
#pragma unroll
  for (int r=0;r<8;r++)
    QK[(size_t)bh*SQ*SQ + (ib+r)*SQ + j] = acc[r]*0.125f;
}

// ---------------- D-matrix gating + stabilized row normalization -> Cn (f16) ----------------
__global__ void attn_norm_kernel(const float* __restrict__ QK, const float* __restrict__ CS,
                                 const float* __restrict__ IG, _Float16* __restrict__ CN){
  int bh = blockIdx.x >> 6; int i = blockIdx.x & 63; int j = threadIdx.x;
  __shared__ float sh[SQ];
  float csI = CS[bh*(SQ+1) + i + 1];
  float ld = (j<=i) ? (csI - CS[bh*(SQ+1) + j + 1] + IG[bh*SQ + j]) : -INFINITY;
  sh[j]=ld; __syncthreads();
  for (int st=32; st>0; st>>=1){ if (j<st) sh[j]=fmaxf(sh[j], sh[j+st]); __syncthreads(); }
  float mx = sh[0]; __syncthreads();
  float dm = (j<=i) ? expf(ld - mx) : 0.f;
  float cm = QK[(size_t)bh*SQ*SQ + i*SQ + j] * dm;
  sh[j]=cm; __syncthreads();
  for (int st=32; st>0; st>>=1){ if (j<st) sh[j]+=sh[j+st]; __syncthreads(); }
  float ssum = sh[0];
  float norm = fmaxf(fabsf(ssum), expf(-mx));
  CN[(size_t)bh*SQ*SQ + i*SQ + j] = (_Float16)(cm/(norm+1e-6f));
}

// ---------------- batched Cn @ v ----------------
__global__ void attn_av_kernel(const _Float16* __restrict__ CN, const float* __restrict__ V,
                               float* __restrict__ H){
  int lane = threadIdx.x & 31;
  int t = blockIdx.x;  int bh = blockIdx.y;
  int b = bh >> 2; int h = bh & 3;
  int mt = t >> 2, nt = t & 3;
  int rowbase = b*SQ; int hoff = h*DHMQ;
  int i_a = mt*16 + (lane&15);
  int dcol = nt*16 + (lane&15);
  v8f acc = {};
  for (int k0=0; k0<SQ; k0+=32){
    int ka = k0 + ((lane&16)?8:0);
    const _Float16* ab = CN + (size_t)bh*SQ*SQ + i_a*SQ + ka;
    v16h a, bv;
#pragma unroll
    for (int e=0;e<8;e++){ a[e]=ab[e]; a[8+e]=ab[16+e]; }
#pragma unroll
    for (int e=0;e<8;e++){
      bv[e]   = (_Float16)V[(size_t)(rowbase + ka + e   )*INNERQ + hoff + dcol];
      bv[8+e] = (_Float16)V[(size_t)(rowbase + ka + 16+e)*INNERQ + hoff + dcol];
    }
    acc = __builtin_amdgcn_wmma_f32_16x16x32_f16(false, a, false, bv, (short)0, acc, false, false);
  }
  int ib = mt*16 + ((lane&16)?8:0);
#pragma unroll
  for (int r=0;r<8;r++)
    H[(size_t)(rowbase + ib + r)*INNERQ + hoff + dcol] = acc[r];
}

// ---------------- per-head LN (hd=64) + skip*conv, * silu(z) ----------------
__global__ void mlstm_combine_kernel(const float* __restrict__ HA, const float* __restrict__ onw,
                                     const float* __restrict__ skip, const float* __restrict__ XC,
                                     const float* __restrict__ U, float* __restrict__ HC){
  int row = blockIdx.x, t = threadIdx.x;
  __shared__ float sh[INNERQ];
  float v = HA[(size_t)row*INNERQ + t];
  const int hd = DHMQ;
  int seg = t & ~(hd-1); int lane = t & (hd-1);
  sh[t]=v; __syncthreads();
  for (int st=hd>>1; st>0; st>>=1){ if (lane<st) sh[t]+=sh[t+st]; __syncthreads(); }
  float mu = sh[seg]/(float)hd; __syncthreads();
  float d = v-mu; sh[t]=d*d; __syncthreads();
  for (int st=hd>>1; st>0; st>>=1){ if (lane<st) sh[t]+=sh[t+st]; __syncthreads(); }
  float var = sh[seg]/(float)hd;
  float yln = d*rsqrtf(var+1e-5f)*onw[t];
  float z   = U[(size_t)row*(2*INNERQ) + INNERQ + t];
  HC[(size_t)row*INNERQ + t] = (yln + skip[t]*XC[(size_t)row*INNERQ + t]) * silu_(z);
}

// ================= host launch =================
extern "C" void kernel_launch(void* const* d_in, const int* in_sizes, int n_in,
                              void* d_out, int out_size, void* d_ws, size_t ws_size,
                              hipStream_t stream) {
  (void)in_sizes; (void)n_in; (void)out_size; (void)ws_size;
  const int*   x        = (const int*)  d_in[0];
  const float* embed    = (const float*)d_in[1];
  const float* s_ln1    = (const float*)d_in[2];
  const float* s_gw     = (const float*)d_in[3];
  const float* s_rec    = (const float*)d_in[4];
  const float* s_bias   = (const float*)d_in[5];
  const float* s_gn     = (const float*)d_in[6];
  const float* s_ln2    = (const float*)d_in[7];
  const float* s_ff_up  = (const float*)d_in[8];
  const float* s_ff_dn  = (const float*)d_in[9];
  const float* m_ln1    = (const float*)d_in[10];
  const float* m_up     = (const float*)d_in[11];
  const float* m_cw     = (const float*)d_in[12];
  const float* m_cb     = (const float*)d_in[13];
  const float* m_qkvw   = (const float*)d_in[14];
  const float* m_igw    = (const float*)d_in[15];
  const float* m_igb    = (const float*)d_in[16];
  const float* m_fgw    = (const float*)d_in[17];
  const float* m_fgb    = (const float*)d_in[18];
  const float* m_skip   = (const float*)d_in[19];
  const float* m_onorm  = (const float*)d_in[20];
  const float* m_down   = (const float*)d_in[21];
  const float* m_ln2    = (const float*)d_in[22];
  const float* m_ff_up  = (const float*)d_in[23];
  const float* m_ff_dn  = (const float*)d_in[24];
  const float* post_ln  = (const float*)d_in[25];
  const float* head_w   = (const float*)d_in[26];
  const float* head_b   = (const float*)d_in[27];
  float* out = (float*)d_out;

  float* ws = (float*)d_ws;
  size_t off = 0;
  auto alloc = [&](size_t n){ float* p = ws + off; off += n; return p; };
  float* H  = alloc((size_t)MROWS*DQ);
  float* A  = alloc((size_t)MROWS*DQ);
  float* Y  = alloc((size_t)MROWS*DQ);
  float* G  = alloc((size_t)MROWS*512);
  float* F1 = alloc((size_t)MROWS*256);
  float* F2 = alloc((size_t)MROWS*128);
  float* XC = alloc((size_t)MROWS*INNERQ);
  float* Qb = alloc((size_t)MROWS*INNERQ);
  float* Kb = alloc((size_t)MROWS*INNERQ);
  float* Vb = alloc((size_t)MROWS*INNERQ);
  float* IG = alloc((size_t)BHQ*SQ);
  float* FG = alloc((size_t)BHQ*SQ);
  float* CS = alloc((size_t)BHQ*(SQ+1));

  embed_kernel<<<MROWS, DQ, 0, stream>>>(x, embed, H);

  for (int l=0; l<2; l++){
    const float* s_ln1_l  = s_ln1  + (size_t)l*DQ;
    const float* s_gw_l   = s_gw   + (size_t)l*4*NHQ*HDQ*HDQ;
    const float* s_rec_l  = s_rec  + (size_t)l*4*NHQ*HDQ*HDQ;
    const float* s_bias_l = s_bias + (size_t)l*4*NHQ*HDQ;
    const float* s_gn_l   = s_gn   + (size_t)l*DQ;
    const float* s_ln2_l  = s_ln2  + (size_t)l*DQ;
    const float* s_fu_l   = s_ff_up+ (size_t)l*DQ*2*FFQ;
    const float* s_fd_l   = s_ff_dn+ (size_t)l*FFQ*DQ;
    const float* m_ln1_l  = m_ln1  + (size_t)l*DQ;
    const float* m_up_l   = m_up   + (size_t)l*DQ*2*INNERQ;
    const float* m_cw_l   = m_cw   + (size_t)l*INNERQ*KCONV;
    const float* m_cb_l   = m_cb   + (size_t)l*INNERQ;
    const float* m_qkv_l  = m_qkvw + (size_t)l*3*NBQ*BSQ*BSQ;
    const float* m_igw_l  = m_igw  + (size_t)l*3*INNERQ*NHMQ;
    const float* m_igb_l  = m_igb  + (size_t)l*NHMQ;
    const float* m_fgw_l  = m_fgw  + (size_t)l*3*INNERQ*NHMQ;
    const float* m_fgb_l  = m_fgb  + (size_t)l*NHMQ;
    const float* m_skip_l = m_skip + (size_t)l*INNERQ;
    const float* m_on_l   = m_onorm+ (size_t)l*INNERQ;
    const float* m_dn_l   = m_down + (size_t)l*INNERQ*DQ;
    const float* m_ln2_l  = m_ln2  + (size_t)l*DQ;
    const float* m_fu_l   = m_ff_up+ (size_t)l*DQ*2*FFQ;
    const float* m_fd_l   = m_ff_dn+ (size_t)l*FFQ*DQ;

    // ---- sLSTM block ----
    ln_kernel<<<MROWS, DQ, 0, stream>>>(H, s_ln1_l, A);
    slstm_gx_kernel<<<(MROWS*512)/256, 256, 0, stream>>>(A, s_gw_l, s_bias_l, G);
    slstm_scan_kernel<<<BQ*NHQ, 128, 0, stream>>>(G, s_rec_l, Y);
    mhln_add_kernel<<<MROWS, DQ, 0, stream>>>(Y, s_gn_l, H, HDQ);

    // ---- FFN 1 ----
    ln_kernel<<<MROWS, DQ, 0, stream>>>(H, s_ln2_l, A);
    gemm_wmma_kernel<false,false><<<dim3(2*FFQ/16, MROWS/64), 128, 0, stream>>>(A, s_fu_l, nullptr, nullptr, F1, MROWS, 2*FFQ, DQ);
    gelu_gate_kernel<<<((size_t)MROWS*FFQ)/256, 256, 0, stream>>>(F1, F2);
    gemm_wmma_kernel<false,true><<<dim3(DQ/16, MROWS/64), 128, 0, stream>>>(F2, s_fd_l, nullptr, H, H, MROWS, DQ, FFQ);

    // ---- mLSTM block ----
    ln_kernel<<<MROWS, DQ, 0, stream>>>(H, m_ln1_l, A);
    gemm_wmma_kernel<false,false><<<dim3(2*INNERQ/16, MROWS/64), 128, 0, stream>>>(A, m_up_l, nullptr, nullptr, G, MROWS, 2*INNERQ, DQ);
    conv_kernel<<<((size_t)MROWS*INNERQ)/256, 256, 0, stream>>>(G, m_cw_l, m_cb_l, XC);
    qkv_kernel<<<dim3(((size_t)MROWS*INNERQ)/256, 3), 256, 0, stream>>>(XC, G, m_qkv_l, Qb, Kb, Vb);
    gates_kernel<<<((size_t)MROWS*NHMQ)/128, 128, 0, stream>>>(Qb, Kb, Vb, m_igw_l, m_igb_l, m_fgw_l, m_fgb_l, IG, FG);
    cumsum_kernel<<<BHQ, 32, 0, stream>>>(FG, CS);
    attn_qk_kernel<<<dim3(16, BHQ), 32, 0, stream>>>(Qb, Kb, F1);
    attn_norm_kernel<<<BHQ*SQ, SQ, 0, stream>>>(F1, CS, IG, (_Float16*)F2);
    attn_av_kernel<<<dim3(16, BHQ), 32, 0, stream>>>((const _Float16*)F2, Vb, Qb);
    mlstm_combine_kernel<<<MROWS, INNERQ, 0, stream>>>(Qb, m_on_l, m_skip_l, XC, G, Kb);
    gemm_wmma_kernel<false,true><<<dim3(DQ/16, MROWS/64), 128, 0, stream>>>(Kb, m_dn_l, nullptr, H, H, MROWS, DQ, INNERQ);

    // ---- FFN 2 ----
    ln_kernel<<<MROWS, DQ, 0, stream>>>(H, m_ln2_l, A);
    gemm_wmma_kernel<false,false><<<dim3(2*FFQ/16, MROWS/64), 128, 0, stream>>>(A, m_fu_l, nullptr, nullptr, F1, MROWS, 2*FFQ, DQ);
    gelu_gate_kernel<<<((size_t)MROWS*FFQ)/256, 256, 0, stream>>>(F1, F2);
    gemm_wmma_kernel<false,true><<<dim3(DQ/16, MROWS/64), 128, 0, stream>>>(F2, m_fd_l, nullptr, H, H, MROWS, DQ, FFQ);
  }

  // ---- final LN + head ----
  ln_kernel<<<MROWS, DQ, 0, stream>>>(H, post_ln, A);
  gemm_wmma_kernel<true,false><<<dim3(VQ/16, MROWS/64), 128, 0, stream>>>(A, head_w, head_b, nullptr, out, MROWS, VQ, DQ);
}
